// TransformerBlock_2353642078630
// MI455X (gfx1250) — matrix operand order repaired and checked by reading, not compiled, not run on hardware
//
#include <hip/hip_runtime.h>

// ---------------------------------------------------------------------------
// MI455X (gfx1250) transformer block.  Compute-bound (~172 GFLOP vs ~200MB of
// HBM traffic at 23.3 TB/s), so every matmul goes through
// v_wmma_f32_16x16x32_bf16 (bf16 A/B, f32 accumulate).  Elementwise work
// (bias / GELU / residual / softmax) is fused into epilogues.
// ---------------------------------------------------------------------------

typedef __bf16 v16bf __attribute__((ext_vector_type(16)));
typedef float  v8f   __attribute__((ext_vector_type(8)));

#define EMB   1024
#define SEQL  2048
#define NHEAD 16
#define HDIM  64
#define ROWS  4096   // BATCH * SEQ

static __device__ __forceinline__ __bf16 f2bf(float f) {
  union { float f; unsigned u; } in; in.f = f;
  unsigned u = in.u;
  u += 0x7fffu + ((u >> 16) & 1u);          // round-to-nearest-even
  union { unsigned short s; __bf16 b; } out;
  out.s = (unsigned short)(u >> 16);
  return out.b;
}

union Frag16 {              // one WMMA 16-bit operand (8 VGPRs / lane)
  v16bf v;
  uint4 q[2];
  __bf16 h[16];
};

// ---------------------------------------------------------------------------
// LayerNorm (unbiased var, ddof=1) -> bf16.  One 256-thread block per row.
// ---------------------------------------------------------------------------
__global__ __launch_bounds__(256)
void ln_bf16(const float* __restrict__ x, const float* __restrict__ sc,
             const float* __restrict__ sh, __bf16* __restrict__ out) {
  __shared__ float r1[256], r2[256];
  const int row = blockIdx.x, tid = threadIdx.x;
  const float* xr = x + (size_t)row * EMB;
  float v[4], s = 0.f, ss = 0.f;
#pragma unroll
  for (int i = 0; i < 4; ++i) {
    v[i] = xr[tid + i * 256];
    s += v[i]; ss += v[i] * v[i];
  }
  r1[tid] = s; r2[tid] = ss;
  __syncthreads();
  for (int st = 128; st > 0; st >>= 1) {
    if (tid < st) { r1[tid] += r1[tid + st]; r2[tid] += r2[tid + st]; }
    __syncthreads();
  }
  const float mean = r1[0] * (1.f / 1024.f);
  const float var  = (r2[0] - 1024.f * mean * mean) * (1.f / 1023.f);
  const float rstd = rsqrtf(var + 1e-5f);
  __bf16* orow = out + (size_t)row * EMB;
#pragma unroll
  for (int i = 0; i < 4; ++i) {
    const int c = tid + i * 256;
    orow[c] = f2bf(sc[c] * (v[i] - mean) * rstd + sh[c]);
  }
}

// ---------------------------------------------------------------------------
// GEMM: C[M,N] = act(A_bf16[M,K] @ W_f32[K,N] + bias) (+res).  Block tile
// 128x64, BK=32, 8 waves -> each wave a 32x32 tile (2x2 WMMA frags).
// Double-buffered LDS: next tile's global loads are issued into registers
// before the barrier so they overlap the 4 WMMAs, then staged into the
// other LDS buffer.  W is converted f32->bf16 while staging transposed so
// B fragments are contiguous 16-half reads (this code's k order: K = e+16*lanehi).
// ---------------------------------------------------------------------------
#define BM 128
#define BN 64
#define BK 32

template <bool GELU_ACT, bool RES, bool OUTBF>
__global__ __launch_bounds__(256)
void gemm_bf16(const __bf16* __restrict__ A, const float* __restrict__ B,
               const float* __restrict__ bias, const float* __restrict__ res,
               void* __restrict__ outv, int M, int N, int K) {
  __shared__ __align__(16) __bf16 sA[2][BM * BK];
  __shared__ __align__(16) __bf16 sBt[2][BN * BK];   // transposed: [n][k]

  const int tid  = threadIdx.x;
  const int lane = tid & 31, wave = tid >> 5;
  const int lm   = lane & 15, lhi = lane >> 4;
  const int wm   = wave >> 1, wn = wave & 1;
  const int m0   = blockIdx.y * BM, n0 = blockIdx.x * BN;

  // Per-thread load coordinates (constant across K-steps).
  const int rowA = tid >> 2, segA = tid & 3;     // +64 rows for second chunk
  const int rowB = tid >> 4, segB = tid & 15;    // +16 rows for second chunk
  const __bf16* aP0 = A + (size_t)(m0 + rowA) * K + segA * 8;
  const __bf16* aP1 = A + (size_t)(m0 + rowA + 64) * K + segA * 8;
  const float*  bP0 = B + (size_t)rowB * N + n0 + segB * 4;
  const float*  bP1 = B + (size_t)(rowB + 16) * N + n0 + segB * 4;

  uint4  ra[2];
  float4 rb[2];
  auto gload = [&](int kt) {
    const int k0 = kt * BK;
    ra[0] = *reinterpret_cast<const uint4*>(aP0 + k0);
    ra[1] = *reinterpret_cast<const uint4*>(aP1 + k0);
    rb[0] = *reinterpret_cast<const float4*>(bP0 + (size_t)k0 * N);
    rb[1] = *reinterpret_cast<const float4*>(bP1 + (size_t)k0 * N);
  };
  auto stage = [&](int buf) {
    *reinterpret_cast<uint4*>(&sA[buf][rowA * BK + segA * 8])        = ra[0];
    *reinterpret_cast<uint4*>(&sA[buf][(rowA + 64) * BK + segA * 8]) = ra[1];
#pragma unroll
    for (int c = 0; c < 2; ++c) {
      const float4 f = rb[c];
      const int nl = segB * 4, kk = rowB + 16 * c;
      sBt[buf][(nl + 0) * BK + kk] = f2bf(f.x);
      sBt[buf][(nl + 1) * BK + kk] = f2bf(f.y);
      sBt[buf][(nl + 2) * BK + kk] = f2bf(f.z);
      sBt[buf][(nl + 3) * BK + kk] = f2bf(f.w);
    }
  };

  const v8f zero = {0.f, 0.f, 0.f, 0.f, 0.f, 0.f, 0.f, 0.f};
  v8f acc[2][2];
  acc[0][0] = zero; acc[0][1] = zero; acc[1][0] = zero; acc[1][1] = zero;

  const int ksteps = K / BK;
  gload(0);
  stage(0);
  for (int kt = 0; kt < ksteps; ++kt) {
    const int cur = kt & 1;
    if (kt + 1 < ksteps) gload(kt + 1);                 // overlap with WMMAs
    if (kt + 2 < ksteps)                                // global_prefetch_b8
      __builtin_prefetch(aP0 + (kt + 2) * BK, 0, 1);
    __syncthreads();                                    // cur buffer ready

    Frag16 fa[2], fb[2];
#pragma unroll
    for (int fm = 0; fm < 2; ++fm) {  // A frag: K = e + 16*lhi (contiguous)
      const int row = wm * 32 + fm * 16 + lm;
      fa[fm].q[0] = *reinterpret_cast<const uint4*>(&sA[cur][row * BK + lhi * 16]);
      fa[fm].q[1] = *reinterpret_cast<const uint4*>(&sA[cur][row * BK + lhi * 16 + 8]);
    }
#pragma unroll
    for (int fn = 0; fn < 2; ++fn) {  // B frag: K = e + 16*lhi (contiguous)
      const int col = wn * 32 + fn * 16 + lm;
      const uint4* p = reinterpret_cast<const uint4*>(&sBt[cur][col * BK + lhi * 16]);
      fb[fn].q[0] = p[0];
      fb[fn].q[1] = p[1];
    }
#pragma unroll
    for (int fm = 0; fm < 2; ++fm)
#pragma unroll
      for (int fn = 0; fn < 2; ++fn)
        acc[fm][fn] = __builtin_amdgcn_wmma_f32_16x16x32_bf16(
            false, fa[fm].v, false, fb[fn].v, (short)0, acc[fm][fn], false, false);

    if (kt + 1 < ksteps) stage((kt + 1) & 1);           // other buffer: no WAR
  }

  // Epilogue: bias (+GELU via sigmoid identity) (+residual), store f32/bf16.
#pragma unroll
  for (int fm = 0; fm < 2; ++fm)
#pragma unroll
    for (int fn = 0; fn < 2; ++fn) {
      const int col = n0 + wn * 32 + fn * 16 + lm;
      const float bv = bias[col];
#pragma unroll
      for (int r = 0; r < 8; ++r) {
        const int mg = m0 + wm * 32 + fm * 16 + r + 8 * lhi;
        float v = acc[fm][fn][r] + bv;
        if (GELU_ACT) {
          // 0.5v(1+tanh(y)) == v * sigmoid(2y),  y = c*(v + 0.044715 v^3)
          const float y2 = 1.5957691216057308f * (v + 0.044715f * v * v * v);
          v = v / (1.f + __expf(-y2));
        }
        const size_t off = (size_t)mg * N + col;
        if (RES) v += res[off];
        if (OUTBF) reinterpret_cast<__bf16*>(outv)[off] = f2bf(v);
        else       reinterpret_cast<float*>(outv)[off]  = v;
      }
    }
}

// ---------------------------------------------------------------------------
// Causal attention, flash-style.  128-thread blocks = 4 independent waves,
// each wave owns one (batch, head, 16-query tile); no block barriers (LDS
// staging is per-wave, guarded by s_wait_dscnt).  32 keys per step: S0/S1
// score tiles via WMMA over D=64, fully-unrolled online softmax with 16-lane
// shfl_xor reductions, P re-laid-out through LDS into an A fragment, PV WMMA.
// ---------------------------------------------------------------------------
__global__ __launch_bounds__(128)
void attn_bf16(const __bf16* __restrict__ Q, const __bf16* __restrict__ Kp,
               const __bf16* __restrict__ Vp, __bf16* __restrict__ O) {
  __shared__ __align__(16) __bf16 sP[4][16 * 32];
  const int tid  = threadIdx.x;
  const int wave = tid >> 5, lane = tid & 31;
  const int lm = lane & 15, lhi = lane >> 4;
  const int qi = blockIdx.x * 4 + wave;
  const int h = blockIdx.y, b = blockIdx.z;
  const int HD = h * HDIM;
  const size_t rowbase = (size_t)b * SEQL;
  __bf16* sPw = sP[wave];
  const v8f zero = {0.f, 0.f, 0.f, 0.f, 0.f, 0.f, 0.f, 0.f};

  // Q fragments for the two 32-wide D chunks (registers for whole loop).
  Frag16 qa[2];
  {
    const __bf16* qrow = Q + (rowbase + qi * 16 + lm) * EMB + HD;
#pragma unroll
    for (int dc = 0; dc < 2; ++dc) {
      qa[dc].q[0] = *reinterpret_cast<const uint4*>(qrow + dc * 32 + 16 * lhi);
      qa[dc].q[1] = *reinterpret_cast<const uint4*>(qrow + dc * 32 + 16 * lhi + 8);
    }
  }

  float m_run[8], l_run[8];
  v8f o[4];
#pragma unroll
  for (int r = 0; r < 8; ++r) { m_run[r] = -1e30f; l_run[r] = 0.f; }
#pragma unroll
  for (int t = 0; t < 4; ++t) o[t] = zero;

  const int nsteps = (qi * 16 + 47) >> 5;   // cover keys 0 .. qi*16+15
  for (int st = 0; st < nsteps; ++st) {
    const int jb = st * 32;

    // ---- S = Q K^T: batch all 4 K fragments, then 4 WMMAs.
    // B frag: key = jb+half*16+lm (lane), d = dc*32 + 16*lhi + e (contig).
    Frag16 kf[2][2];
#pragma unroll
    for (int half = 0; half < 2; ++half) {
      const __bf16* krow = Kp + (rowbase + jb + half * 16 + lm) * EMB + HD;
#pragma unroll
      for (int dc = 0; dc < 2; ++dc) {
        kf[half][dc].q[0] = *reinterpret_cast<const uint4*>(krow + dc * 32 + 16 * lhi);
        kf[half][dc].q[1] = *reinterpret_cast<const uint4*>(krow + dc * 32 + 16 * lhi + 8);
      }
    }
    v8f s0 = zero, s1 = zero;
#pragma unroll
    for (int dc = 0; dc < 2; ++dc) {
      s0 = __builtin_amdgcn_wmma_f32_16x16x32_bf16(false, qa[dc].v, false,
                                                   kf[0][dc].v, (short)0, s0, false, false);
      s1 = __builtin_amdgcn_wmma_f32_16x16x32_bf16(false, qa[dc].v, false,
                                                   kf[1][dc].v, (short)0, s1, false, false);
    }

    // ---- Scale, causal mask, online softmax (rows across 16 lanes/half).
    float p0[8], p1[8], alpha[8];
#pragma unroll
    for (int r = 0; r < 8; ++r) {
      const int gq = qi * 16 + r + 8 * lhi;
      float a0 = s0[r] * 0.125f;   // 1/sqrt(64)
      float a1 = s1[r] * 0.125f;
      if (jb + lm > gq)      a0 = -1e30f;
      if (jb + 16 + lm > gq) a1 = -1e30f;
      float rmax = fmaxf(a0, a1);
      rmax = fmaxf(rmax, __shfl_xor(rmax, 1, 32));
      rmax = fmaxf(rmax, __shfl_xor(rmax, 2, 32));
      rmax = fmaxf(rmax, __shfl_xor(rmax, 4, 32));
      rmax = fmaxf(rmax, __shfl_xor(rmax, 8, 32));
      const float nm = fmaxf(m_run[r], rmax);
      alpha[r] = __expf(m_run[r] - nm);
      m_run[r] = nm;
      p0[r] = __expf(a0 - nm);
      p1[r] = __expf(a1 - nm);
      float ps = p0[r] + p1[r];
      ps += __shfl_xor(ps, 1, 32);
      ps += __shfl_xor(ps, 2, 32);
      ps += __shfl_xor(ps, 4, 32);
      ps += __shfl_xor(ps, 8, 32);
      l_run[r] = l_run[r] * alpha[r] + ps;
    }
#pragma unroll
    for (int t = 0; t < 4; ++t)
#pragma unroll
      for (int r = 0; r < 8; ++r) o[t][r] *= alpha[r];

    // ---- Stage P (16q x 32k bf16) in this wave's LDS slice, reload as A frag.
#pragma unroll
    for (int r = 0; r < 8; ++r) {
      const int m = r + 8 * lhi;
      sPw[m * 32 + lm]      = f2bf(p0[r]);
      sPw[m * 32 + 16 + lm] = f2bf(p1[r]);
    }
    __asm__ volatile("s_wait_dscnt 0x0" ::: "memory");  // intra-wave LDS RAW
    Frag16 pf;
    pf.q[0] = *reinterpret_cast<const uint4*>(&sPw[lm * 32 + 16 * lhi]);
    pf.q[1] = *reinterpret_cast<const uint4*>(&sPw[lm * 32 + 16 * lhi + 8]);

    // ---- O += P @ V: batch all 4 V fragments, then 4 WMMAs.
    // V B-frag: d = t*16+lm (lane), key = jb + 16*lhi + e (stride EMB).
    Frag16 vf[4];
#pragma unroll
    for (int t = 0; t < 4; ++t) {
      const __bf16* vp = Vp + (rowbase + jb + 16 * lhi) * EMB + HD + t * 16 + lm;
#pragma unroll
      for (int e = 0; e < 16; ++e) vf[t].h[e] = vp[(size_t)e * EMB];
    }
#pragma unroll
    for (int t = 0; t < 4; ++t)
      o[t] = __builtin_amdgcn_wmma_f32_16x16x32_bf16(false, pf.v, false, vf[t].v,
                                                     (short)0, o[t], false, false);
  }

  float inv[8];
#pragma unroll
  for (int r = 0; r < 8; ++r) inv[r] = 1.f / l_run[r];
#pragma unroll
  for (int t = 0; t < 4; ++t)
#pragma unroll
    for (int r = 0; r < 8; ++r) {
      const size_t off =
          (rowbase + qi * 16 + r + 8 * lhi) * EMB + HD + t * 16 + lm;
      O[off] = f2bf(o[t][r] * inv[r]);
    }
}

// ---------------------------------------------------------------------------
// Launch.  Workspace (>= 56 MB):
//   [ 0, 8)MB  h  (bf16, LN1 out; reused for LN2 out)
//   [ 8,16)MB  q  (bf16)      \
//   [16,24)MB  k  (bf16)       | dead after O-proj -> region reused as
//   [24,32)MB  v  (bf16)       | FFN hidden f1 (bf16, 32 MB at offset 8MB)
//   [32,40)MB  ctx(bf16)      /
//   [40,56)MB  x1 (f32, post-attention residual)
// ---------------------------------------------------------------------------
extern "C" void kernel_launch(void* const* d_in, const int* in_sizes, int n_in,
                              void* d_out, int out_size, void* d_ws, size_t ws_size,
                              hipStream_t stream) {
  (void)in_sizes; (void)n_in; (void)out_size; (void)ws_size;
  const float* x    = (const float*)d_in[0];
  const float* Wq   = (const float*)d_in[1];
  const float* bq   = (const float*)d_in[2];
  const float* Wk   = (const float*)d_in[3];
  const float* bk   = (const float*)d_in[4];
  const float* Wv   = (const float*)d_in[5];
  const float* bv   = (const float*)d_in[6];
  const float* Wo   = (const float*)d_in[7];
  const float* bo   = (const float*)d_in[8];
  const float* W1   = (const float*)d_in[9];
  const float* b1   = (const float*)d_in[10];
  const float* W2   = (const float*)d_in[11];
  const float* b2   = (const float*)d_in[12];
  const float* ln1s = (const float*)d_in[13];
  const float* ln1h = (const float*)d_in[14];
  const float* ln2s = (const float*)d_in[15];
  const float* ln2h = (const float*)d_in[16];

  char* ws = (char*)d_ws;
  const size_t MBy = (size_t)1 << 20;
  __bf16* hb = (__bf16*)(ws + 0 * MBy);
  __bf16* qb = (__bf16*)(ws + 8 * MBy);
  __bf16* kb = (__bf16*)(ws + 16 * MBy);
  __bf16* vb = (__bf16*)(ws + 24 * MBy);
  __bf16* cb = (__bf16*)(ws + 32 * MBy);
  float*  x1 = (float*)(ws + 40 * MBy);
  __bf16* f1 = (__bf16*)(ws + 8 * MBy);   // reuse q/k/v/ctx region

  const int M = ROWS, E = EMB, F = 4 * EMB;
  const dim3 blk(256);
  const dim3 gE(E / BN, M / BM);
  const dim3 gF(F / BN, M / BM);

  // LN1 -> h
  ln_bf16<<<M, blk, 0, stream>>>(x, ln1s, ln1h, hb);
  // Q, K, V projections (bf16 out)
  gemm_bf16<false, false, true><<<gE, blk, 0, stream>>>(hb, Wq, bq, nullptr, qb, M, E, E);
  gemm_bf16<false, false, true><<<gE, blk, 0, stream>>>(hb, Wk, bk, nullptr, kb, M, E, E);
  gemm_bf16<false, false, true><<<gE, blk, 0, stream>>>(hb, Wv, bv, nullptr, vb, M, E, E);
  // Causal attention -> ctx (bf16)
  attn_bf16<<<dim3(SEQL / 64, NHEAD, 2), dim3(128), 0, stream>>>(qb, kb, vb, cb);
  // O projection + residual(x) -> x1 (f32)
  gemm_bf16<false, true, false><<<gE, blk, 0, stream>>>(cb, Wo, bo, x, x1, M, E, E);
  // LN2 -> h (reuse)
  ln_bf16<<<M, blk, 0, stream>>>(x1, ln2s, ln2h, hb);
  // FFN1 + GELU -> f1 (bf16)
  gemm_bf16<true, false, true><<<gF, blk, 0, stream>>>(hb, W1, b1, nullptr, f1, M, F, E);
  // FFN2 + residual(x1) -> out (f32)
  gemm_bf16<false, true, false><<<gE, blk, 0, stream>>>(f1, W2, b2, x1, (float*)d_out, M, E, F);
}
